// GraphTransformerModel_64209761075680
// MI455X (gfx1250) — matrix operand order repaired and checked by reading, not compiled, not run on hardware
//
#include <hip/hip_runtime.h>
#include <hip/hip_bf16.h>

// ---------------- problem constants ----------------
#define NNODES 50000
#define NEDGES 800000
#define FDIM   128
#define NHEADS 4
#define NGRAPH 64
#define YSTRIDE ((size_t)NNODES * FDIM)   // floats between Q,K,V,OUT planes

#if __has_builtin(__builtin_amdgcn_global_load_async_to_lds_b128)
#define HAVE_ASYNC_LDS 1
#else
#define HAVE_ASYNC_LDS 0
#endif

// ---------------- types for WMMA / async ----------------
typedef __attribute__((ext_vector_type(16))) __bf16        v16bf;
typedef __attribute__((ext_vector_type(8)))  float         v8f;
typedef __attribute__((ext_vector_type(8)))  unsigned int  v8u;
typedef int v4i __attribute__((vector_size(16)));          // pointee type of async builtin

// float -> bf16 (round-to-nearest-even), as raw ushort
__device__ __forceinline__ unsigned int f2bfbits(float f) {
  unsigned int u = __float_as_uint(f);
  u += 0x7FFFu + ((u >> 16) & 1u);
  return u >> 16;
}
__device__ __forceinline__ unsigned short f2bf(float f) {
  return (unsigned short)f2bfbits(f);
}
// monotonic uint encoding of float for atomicMax-based segment max
__device__ __forceinline__ unsigned int fkey(float f) {
  unsigned int u = __float_as_uint(f);
  return (u & 0x80000000u) ? ~u : (u | 0x80000000u);
}
__device__ __forceinline__ float kdec(unsigned int k) {
  unsigned int u = (k & 0x80000000u) ? (k & 0x7FFFFFFFu) : ~k;
  return __uint_as_float(u);
}

// ---------------- weight convert: W[fi][fo] f32 -> Wt[o][n][k] bf16 ----------------
__global__ __launch_bounds__(256) void wconv_kernel(
    const float* __restrict__ Wq, const float* __restrict__ Wk,
    const float* __restrict__ Wv, const float* __restrict__ Ws,
    unsigned short* __restrict__ Wt) {
  int i = blockIdx.x * 256 + threadIdx.x;          // 0 .. 4*128*128-1
  int o = i >> 14;
  int n = (i >> 7) & 127;
  int k = i & 127;
  const float* W = (o == 0) ? Wq : (o == 1) ? Wk : (o == 2) ? Wv : Ws;
  Wt[i] = f2bf(W[k * FDIM + n]);
}

// ---------------- stage 4 bias vectors contiguously: Bc[o*128+n] ----------------
__global__ __launch_bounds__(256) void bias_stage_kernel(
    const float* __restrict__ b0, const float* __restrict__ b1,
    const float* __restrict__ b2, const float* __restrict__ b3,
    float* __restrict__ Bc) {
  int i = blockIdx.x * 256 + threadIdx.x;          // 0..511
  if (i < 4 * FDIM) {
    int o = i >> 7, n = i & 127;
    const float* b = (o == 0) ? b0 : (o == 1) ? b1 : (o == 2) ? b2 : b3;
    Bc[i] = b[n];
  }
}

// ---------------- activation f32 -> bf16 plane ----------------
__global__ void x2bf_kernel(const float* __restrict__ in,
                            unsigned short* __restrict__ outb, int n) {
  int i = blockIdx.x * 256 + threadIdx.x;
  if (i < n) outb[i] = f2bf(in[i]);
}

// ---------------- ReLU (in-place f32) fused with bf16 conversion ----------------
__global__ void relu_bf_kernel(float* __restrict__ io,
                               unsigned short* __restrict__ outb, int n) {
  int i = blockIdx.x * 256 + threadIdx.x;
  if (i < n) {
    float r = fmaxf(io[i], 0.0f);
    io[i]   = r;                                   // f32 plane (pooling / next stages)
    outb[i] = f2bf(r);                             // bf16 plane (next layer's GEMM)
  }
}

// ---------------- fused 4-output GEMM: Y_o = X @ W_o + b_o ----------------
// block = 256 threads = 8 waves; block covers 32 rows of X (bf16 plane).
// wave w: row-tile rt = w>>2 (0/1), output o = w&3 (Q,K,V,S->OUT).
// X tile staged bf16 into LDS via async-load-to-LDS (raw copy, no VALU);
// A-fragments are pure ds_load_b128. K-step outer / col-tile inner with 8
// live accumulators -> independent v_wmma, no hazard v_nops.
__global__ __launch_bounds__(256) void gemm4_kernel(
    const unsigned short* __restrict__ Xb, const unsigned short* __restrict__ Wt,
    const float* __restrict__ Bc, float* __restrict__ Qbase, int nrows) {
  __shared__ __align__(16) unsigned short xt[32][136];   // 32 rows x 128 cols bf16, padded

  const int tid  = threadIdx.x;
  const int row0 = blockIdx.x * 32;

  const int wave = tid >> 5;
  const int lane = tid & 31;
  const int rt   = wave >> 2;                      // row tile 0/1
  const int o    = wave & 3;                       // which output
  const float* bias = Bc + o * FDIM;
  float*       Y    = Qbase + (size_t)o * YSTRIDE;
  const unsigned short* W = Wt + (size_t)o * FDIM * FDIM;   // Wt[o][n][k]

  // keep this wave's 32KB weight panel hot in cache (global_prefetch_b8)
  __builtin_prefetch((const char*)W + (size_t)lane * 512, 0, 3);

  // ---- stage X tile into LDS (each thread: 1 row x 16 halves = 2x b128) ----
  {
    int r  = tid >> 3;
    int c0 = (tid & 7) * 16;                       // halves
    int grow = row0 + r;
    if (grow > nrows - 1) grow = nrows - 1;        // clamp (stores are guarded later)
    const unsigned short* gsrc = Xb + (size_t)grow * FDIM + c0;
    unsigned short* lds = &xt[r][c0];
#if HAVE_ASYNC_LDS
#pragma unroll
    for (int j = 0; j < 2; ++j) {
      __builtin_amdgcn_global_load_async_to_lds_b128(
          (__attribute__((address_space(1))) v4i*)(gsrc + j * 8),
          (__attribute__((address_space(3))) v4i*)(lds + j * 8), 0, 0);
    }
#if __has_builtin(__builtin_amdgcn_s_wait_asynccnt)
    __builtin_amdgcn_s_wait_asynccnt(0);
#else
    asm volatile("s_wait_asynccnt 0" ::: "memory");
#endif
#else
#pragma unroll
    for (int j = 0; j < 2; ++j) {
      *(uint4*)(lds + j * 8) = *(const uint4*)(gsrc + j * 8);
    }
#endif
  }
  __syncthreads();

  // ---- A fragments (16x32 bf16): lane<16 -> K {k0..k0+7, k0+16..k0+23},
  //                                lane>=16 -> K {k0+8..k0+15, k0+24..k0+31}
  const int arow  = rt * 16 + (lane & 15);
  const int abase = (lane < 16) ? 0 : 8;
  v16bf afrag[4];
#pragma unroll
  for (int ks = 0; ks < 4; ++ks) {
    int k0 = ks * 32;
    uint4 lo = *(const uint4*)&xt[arow][k0 + abase];
    uint4 hi = *(const uint4*)&xt[arow][k0 + abase + 16];
    v8u au = {lo.x, lo.y, lo.z, lo.w, hi.x, hi.y, hi.z, hi.w};
    afrag[ks] = __builtin_bit_cast(v16bf, au);
  }

  // ---- main loop: 8 independent accumulators, K-step outer ----
  const int nrel = lane & 15;
  const int kb   = (lane < 16) ? 0 : 8;
  v8f acc[8];
#pragma unroll
  for (int ct = 0; ct < 8; ++ct) acc[ct] = (v8f){};

#pragma unroll
  for (int ks = 0; ks < 4; ++ks) {
    v16bf bfr[8];
#pragma unroll
    for (int ct = 0; ct < 8; ++ct) {
      const uint4* bp =
          (const uint4*)(W + (size_t)(ct * 16 + nrel) * FDIM + ks * 32 + kb);
      uint4 b0 = bp[0];
      uint4 b1 = bp[2];
      v8u bu = {b0.x, b0.y, b0.z, b0.w, b1.x, b1.y, b1.z, b1.w};
      bfr[ct] = __builtin_bit_cast(v16bf, bu);
    }
#pragma unroll
    for (int ct = 0; ct < 8; ++ct) {
      acc[ct] = __builtin_amdgcn_wmma_f32_16x16x32_bf16(
          false, afrag[ks], false, bfr[ct], (short)0, acc[ct], false, false);
    }
  }

  // ---- epilogue: branch-free fast path for full blocks ----
  const int  mbase = row0 + rt * 16 + ((lane < 16) ? 0 : 8);
  const bool full  = (row0 + 32 <= nrows);
#pragma unroll
  for (int ct = 0; ct < 8; ++ct) {
    const int   col = ct * 16 + nrel;
    const float bb  = bias[col];
    float* yp = Y + (size_t)mbase * FDIM + col;
    if (full) {
#pragma unroll
      for (int i = 0; i < 8; ++i) yp[(size_t)i * FDIM] = acc[ct][i] + bb;
    } else {
#pragma unroll
      for (int i = 0; i < 8; ++i)
        if (mbase + i < nrows) yp[(size_t)i * FDIM] = acc[ct][i] + bb;
    }
  }
}

// ---------------- per-layer segment-softmax init ----------------
__global__ void seg_init_kernel(unsigned int* __restrict__ mkey,
                                float* __restrict__ denom, int n) {
  int i = blockIdx.x * blockDim.x + threadIdx.x;
  if (i < n) { mkey[i] = 0x007FFFFFu /* encode(-inf) */; denom[i] = 0.0f; }
}

// ---------------- edge logits + segment max (one wave per edge) ----------------
__global__ __launch_bounds__(256) void edge_logits_kernel(
    const int* __restrict__ src, const int* __restrict__ dst,
    const float* __restrict__ Q, const float* __restrict__ K,
    float* __restrict__ alpha, unsigned int* __restrict__ mkey, int ne) {
  int wid  = (blockIdx.x * 256 + threadIdx.x) >> 5;
  int lane = threadIdx.x & 31;
  if (wid >= ne) return;
  int s = src[wid], d = dst[wid];
  float4 qv = ((const float4*)(Q + (size_t)d * FDIM))[lane];
  float4 kv = ((const float4*)(K + (size_t)s * FDIM))[lane];
  float p = qv.x * kv.x + qv.y * kv.y + qv.z * kv.z + qv.w * kv.w;
  p += __shfl_xor(p, 1, 32);
  p += __shfl_xor(p, 2, 32);
  p += __shfl_xor(p, 4, 32);                       // 8-lane group == one head (32 dims)
  if ((lane & 7) == 0) {
    int h = lane >> 3;
    float a = p * 0.17677669529663689f;            // 1/sqrt(32)
    alpha[(size_t)wid * NHEADS + h] = a;
    atomicMax(&mkey[(size_t)d * NHEADS + h], fkey(a));
  }
}

// ---------------- exp(alpha - max) + segment sum ----------------
__global__ __launch_bounds__(256) void edge_exp_kernel(
    float* __restrict__ alpha, const int* __restrict__ dst,
    const unsigned int* __restrict__ mkey, float* __restrict__ denom, int total) {
  int i = blockIdx.x * 256 + threadIdx.x;
  if (i >= total) return;
  int e = i >> 2, h = i & 3;
  int d = dst[e];
  float ex = __expf(alpha[i] - kdec(mkey[d * NHEADS + h]));
  alpha[i] = ex;
  unsafeAtomicAdd(&denom[d * NHEADS + h], ex);
}

// ---------------- weighted scatter-aggregate (one wave per edge) ----------------
__global__ __launch_bounds__(256) void edge_agg_kernel(
    const int* __restrict__ src, const int* __restrict__ dst,
    const float* __restrict__ alpha, const float* __restrict__ denom,
    const float* __restrict__ V, float* __restrict__ OUT, int ne) {
  int wid  = (blockIdx.x * 256 + threadIdx.x) >> 5;
  int lane = threadIdx.x & 31;
  if (wid >= ne) return;
  int s = src[wid], d = dst[wid];
  int h = lane >> 3;
  float a = alpha[(size_t)wid * NHEADS + h] / denom[d * NHEADS + h];
  float4 vv = ((const float4*)(V + (size_t)s * FDIM))[lane];
  float* op = OUT + (size_t)d * FDIM + lane * 4;
  unsafeAtomicAdd(op + 0, a * vv.x);
  unsafeAtomicAdd(op + 1, a * vv.y);
  unsafeAtomicAdd(op + 2, a * vv.z);
  unsafeAtomicAdd(op + 3, a * vv.w);
}

// ---------------- pooling ----------------
__global__ void pool_zero_kernel(float* __restrict__ sums, float* __restrict__ cnts) {
  int i = blockIdx.x * 256 + threadIdx.x;
  if (i < NGRAPH * FDIM) sums[i] = 0.0f;
  if (i < NGRAPH) cnts[i] = 0.0f;
}

__global__ __launch_bounds__(256) void pool_accum_kernel(
    const int* __restrict__ batch, const float* __restrict__ Hf,
    float* __restrict__ sums, float* __restrict__ cnts, int nn) {
  int wid  = (blockIdx.x * 256 + threadIdx.x) >> 5;
  int lane = threadIdx.x & 31;
  if (wid >= nn) return;
  int g = batch[wid];
  float4 hv = ((const float4*)(Hf + (size_t)wid * FDIM))[lane];
  float* sp = sums + (size_t)g * FDIM + lane * 4;
  unsafeAtomicAdd(sp + 0, hv.x);
  unsafeAtomicAdd(sp + 1, hv.y);
  unsafeAtomicAdd(sp + 2, hv.z);
  unsafeAtomicAdd(sp + 3, hv.w);
  if (lane == 0) unsafeAtomicAdd(&cnts[g], 1.0f);
}

__global__ __launch_bounds__(32) void fc_kernel(
    const float* __restrict__ sums, const float* __restrict__ cnts,
    const float* __restrict__ Wfc, const float* __restrict__ bfc,
    float* __restrict__ out) {
  int g = blockIdx.x, lane = threadIdx.x;
  float4 sv = ((const float4*)(sums + (size_t)g * FDIM))[lane];
  float4 wv = ((const float4*)Wfc)[lane];
  float p = sv.x * wv.x + sv.y * wv.y + sv.z * wv.z + sv.w * wv.w;
  p += __shfl_xor(p, 1, 32);
  p += __shfl_xor(p, 2, 32);
  p += __shfl_xor(p, 4, 32);
  p += __shfl_xor(p, 8, 32);
  p += __shfl_xor(p, 16, 32);
  if (lane == 0) out[g] = p / fmaxf(cnts[g], 1.0f) + bfc[0];
}

// ---------------- host orchestration ----------------
extern "C" void kernel_launch(void* const* d_in, const int* in_sizes, int n_in,
                              void* d_out, int out_size, void* d_ws, size_t ws_size,
                              hipStream_t stream) {
  (void)in_sizes; (void)n_in; (void)out_size; (void)ws_size;

  const float* x     = (const float*)d_in[0];
  const int*   ei    = (const int*)d_in[1];
  const int*   batch = (const int*)d_in[2];
  const int*   srcp  = ei;
  const int*   dstp  = ei + NEDGES;
  const float* Wfc   = (const float*)d_in[19];
  const float* bfc   = (const float*)d_in[20];

  // workspace layout (bytes, 256B aligned); Q,K,V,OUT contiguous planes.
  char* w = (char*)d_ws;
  unsigned short* XBF   = (unsigned short*)(w + 0);     // 12,800,000 (bf16 activations)
  float*          Qb    = (float*)(w + 25600000);
  float*          Vb    = (float*)(w + 76800000);
  float*          OUTb  = (float*)(w + 102400000);
  float*          ALPHA = (float*)(w + 128000000);      // 12,800,000
  unsigned int*   MKEY  = (unsigned int*)(w + 140800000);
  float*          DENOM = (float*)(w + 141600000);
  unsigned short* WT0   = (unsigned short*)(w + 142400000);
  unsigned short* WT1   = (unsigned short*)(w + 142531072);
  float*          SUMS  = (float*)(w + 142662144);
  float*          CNTS  = (float*)(w + 142694912);
  float*          BC0   = (float*)(w + 142695168);      // 4*128 floats
  float*          BC1   = (float*)(w + 142697216);
  float*          Kb    = Qb + YSTRIDE;

  // convert both layers' weights to transposed bf16; stage biases contiguously
  wconv_kernel<<<256, 256, 0, stream>>>((const float*)d_in[3], (const float*)d_in[5],
                                        (const float*)d_in[7], (const float*)d_in[9], WT0);
  wconv_kernel<<<256, 256, 0, stream>>>((const float*)d_in[11], (const float*)d_in[13],
                                        (const float*)d_in[15], (const float*)d_in[17], WT1);
  bias_stage_kernel<<<2, 256, 0, stream>>>((const float*)d_in[4], (const float*)d_in[6],
                                           (const float*)d_in[8], (const float*)d_in[10], BC0);
  bias_stage_kernel<<<2, 256, 0, stream>>>((const float*)d_in[12], (const float*)d_in[14],
                                           (const float*)d_in[16], (const float*)d_in[18], BC1);

  const int nelem    = NNODES * FDIM;                   // 6,400,000
  const int gemmGrid = (NNODES + 31) / 32;              // 1563
  const int segN     = NNODES * NHEADS;                 // 200000
  const int waveGrid = NEDGES / 8;                      // 100000 blocks, 8 waves each

  // layer-0 input: bf16 copy of x
  x2bf_kernel<<<(nelem + 255) / 256, 256, 0, stream>>>(x, XBF, nelem);

  for (int layer = 0; layer < 2; ++layer) {
    gemm4_kernel<<<gemmGrid, 256, 0, stream>>>(
        XBF, layer ? WT1 : WT0, layer ? BC1 : BC0, Qb, NNODES);
    seg_init_kernel<<<(segN + 255) / 256, 256, 0, stream>>>(MKEY, DENOM, segN);
    edge_logits_kernel<<<waveGrid, 256, 0, stream>>>(srcp, dstp, Qb, Kb, ALPHA, MKEY, NEDGES);
    edge_exp_kernel<<<(NEDGES * NHEADS + 255) / 256, 256, 0, stream>>>(
        ALPHA, dstp, MKEY, DENOM, NEDGES * NHEADS);
    edge_agg_kernel<<<waveGrid, 256, 0, stream>>>(srcp, dstp, ALPHA, DENOM, Vb, OUTb, NEDGES);
    // ReLU in-place on f32 plane + refresh bf16 plane for next layer's GEMM
    relu_bf_kernel<<<(nelem + 255) / 256, 256, 0, stream>>>(OUTb, XBF, nelem);
  }

  pool_zero_kernel<<<32, 256, 0, stream>>>(SUMS, CNTS);
  pool_accum_kernel<<<(NNODES + 7) / 8, 256, 0, stream>>>(batch, OUTb, SUMS, CNTS, NNODES);
  fc_kernel<<<NGRAPH, 32, 0, stream>>>(SUMS, CNTS, Wfc, bfc, (float*)d_out);
}
